// PairEmbed_13408887898528
// MI455X (gfx1250) — matrix operand order repaired and checked by reading, not compiled, not run on hardware
//
#include <hip/hip_runtime.h>
#include <hip/hip_bf16.h>
#include <math.h>

// Problem constants (match reference)
#define LNODES 128
#define BATCH  8
#define DEMB   256
#define HDIM   256
#define NHEADS 8
#define NGAUSS 50

typedef __attribute__((ext_vector_type(16))) _Float16 v16h;
typedef __attribute__((ext_vector_type(8)))  _Float16 v8h;
typedef __attribute__((ext_vector_type(8)))  float    v8f;

// ---------- helpers ----------

__device__ __forceinline__ v16h ld_frag(const _Float16* lo, const _Float16* hi) {
  v8h a = *(const v8h*)lo;   // 16B LDS load
  v8h b = *(const v8h*)hi;   // 16B LDS load
  return __builtin_shufflevector(a, b, 0,1,2,3,4,5,6,7,8,9,10,11,12,13,14,15);
}

__device__ __forceinline__ v8f wmma16(v16h a, v16h b, v8f c) {
  // v_wmma_f32_16x16x32_f16
  return __builtin_amdgcn_wmma_f32_16x16x32_f16(false, a, false, b, (short)0, c, false, false);
}

// jax.nn.gelu default (tanh approx): 0.5x(1+tanh(z)) = x * (1 - 1/(exp(2z)+1))
// one v_exp_f32 + one v_rcp_f32; exact at +/-inf (rcp(inf)=0 -> x ; e=0 -> 0)
__device__ __forceinline__ float gelu_tanh(float x) {
  const float c0 = 0.7978845608028654f;   // sqrt(2/pi)
  const float c1 = 0.044715f;
  float z = c0 * (x + c1 * x * x * x);
  float e = __expf(2.0f * z);
  float r = __builtin_amdgcn_rcpf(e + 1.0f);
  return x - x * r;
}

// ---------- prep kernels (tiny vs main work) ----------

// WcT[h][g] = sum_d W_rbf[g][d] * W3[d][h]  (g<50, padded to 64), f16, K-contiguous
// bc[h] = b_in[h] + sum_d b_rbf[d]*W3[d][h]
__global__ void prep_wc(const float* __restrict__ W_rbf, const float* __restrict__ W3,
                        const float* __restrict__ b_rbf, const float* __restrict__ b_in,
                        _Float16* __restrict__ WcT, float* __restrict__ bc) {
  int idx = blockIdx.x * blockDim.x + threadIdx.x;      // 256*64
  if (idx >= 256 * 64) return;
  int h = idx >> 6;
  int g = idx & 63;
  float s = 0.0f;
  if (g < NGAUSS) {
    for (int d = 0; d < DEMB; ++d) s += W_rbf[g * DEMB + d] * W3[d * HDIM + h];
  }
  WcT[h * 64 + g] = (_Float16)s;
  if (g == 0) {
    float s2 = b_in[h];
    for (int d = 0; d < DEMB; ++d) s2 += b_rbf[d] * W3[d * HDIM + h];
    bc[h] = s2;
  }
}

// Transposed f16 weights, K-contiguous per output column:
// WresT[n][k] = W_res[k][n] ; WoutT[n][k] = W_out[k][n] (n<8 else 0, padded to 16 cols)
__global__ void prep_wT(const float* __restrict__ W_res, const float* __restrict__ W_out,
                        _Float16* __restrict__ WresT, _Float16* __restrict__ WoutT) {
  int idx = blockIdx.x * blockDim.x + threadIdx.x;      // 65536
  if (idx < HDIM * HDIM) {
    int n = idx >> 8, k = idx & 255;
    WresT[n * HDIM + k] = (_Float16)W_res[k * HDIM + n];
  }
  if (idx < 16 * HDIM) {
    int n = idx >> 8, k = idx & 255;
    WoutT[n * HDIM + k] = (n < NHEADS) ? (_Float16)W_out[k * NHEADS + n] : (_Float16)0.0f;
  }
}

// aProj[r][h] = x[r]@W1 ; bProj[r][h] = x[r]@W2 ; r = l*B+b  (f32, reused L times each)
__global__ void prep_proj(const float* __restrict__ x, const float* __restrict__ W1,
                          const float* __restrict__ W2,
                          float* __restrict__ aP, float* __restrict__ bP) {
  int idx = blockIdx.x * blockDim.x + threadIdx.x;      // 262144
  if (idx >= LNODES * BATCH * HDIM) return;
  int r = idx >> 8, h = idx & 255;
  const float* xr = x + (size_t)r * DEMB;
  float sa = 0.0f, sb = 0.0f;
  for (int d = 0; d < DEMB; ++d) {
    float xv = xr[d];
    sa += xv * W1[d * HDIM + h];
    sb += xv * W2[d * HDIM + h];
  }
  aP[idx] = sa;
  bP[idx] = sb;
}

// ---------- main fused pair kernel ----------
// One wave handles a 16x(H) tile: fixed (i, b), 16 consecutive j.
// LDS (dynamic, halves): WcT[256][64] @0, WresT[256][256] @16384,
//                        WoutT[16][256] @81920, stage[wave][16][256] @86016 (8 waves)
#define OFF_WC 0
#define OFF_WR 16384
#define OFF_WO 81920
#define OFF_ST 86016
#define LDS_BYTES 237568   // (86016 + 8*4096) halves * 2B = 232 KB  (< 320 KB/WG)

__global__ void __launch_bounds__(256, 1)
pair_main(const float* __restrict__ dist, const unsigned char* __restrict__ mask,
          const float* __restrict__ aProj, const float* __restrict__ bProj,
          const float* __restrict__ bc, const _Float16* __restrict__ gWcT,
          const _Float16* __restrict__ gWresT, const _Float16* __restrict__ gWoutT,
          const float* __restrict__ b_res, const float* __restrict__ b_out,
          float* __restrict__ out) {
  extern __shared__ _Float16 smem[];
  _Float16* sWc = smem + OFF_WC;
  _Float16* sWr = smem + OFF_WR;
  _Float16* sWo = smem + OFF_WO;

  const int tid = threadIdx.x;

  // cooperative b128 fill of shared weights
  {
    uint4* d0 = (uint4*)sWc; const uint4* s0 = (const uint4*)gWcT;
    for (int c = tid; c < 2048; c += 256) d0[c] = s0[c];
    uint4* d1 = (uint4*)sWr; const uint4* s1 = (const uint4*)gWresT;
    for (int c = tid; c < 8192; c += 256) d1[c] = s1[c];
    uint4* d2 = (uint4*)sWo; const uint4* s2 = (const uint4*)gWoutT;
    for (int c = tid; c < 512; c += 256) d2[c] = s2[c];
  }
  __syncthreads();

  const int wave = tid >> 5;
  const int lane = tid & 31;
  const int half = lane >> 4;   // lane group (ISA A/B/C layouts split at lane 16)
  const int l16  = lane & 15;
  _Float16* stage = smem + OFF_ST + wave * (16 * 256);

  const int t  = blockIdx.x * 8 + wave;        // tile id, 8192 total
  const int jt = t & 7;
  const int b  = (t >> 3) & 7;
  const int i  = t >> 6;                       // 0..127
  const int j0 = jt << 4;

  const float DELTA = 12.0f / 49.0f;
  const float COEFF = -0.5f / (DELTA * DELTA);

  // ---- Gaussians directly in WMMA A-fragment layout (16x64, K padded 50->64) ----
  // A f16 16x32, contiguous k order used here (B matches): lane -> m = lane%16 ; element e -> k = (lane/16)*16 + e
  float d = dist[((size_t)i * LNODES + (j0 + l16)) * BATCH + b];
  v16h ag0, ag1;
#pragma unroll
  for (int e = 0; e < 16; ++e) {
    int k0 = (half << 4) + e;   // 0..31 (< NGAUSS always)
    int k1 = 32 + k0;                                   // 32..63
    float t0 = d - (float)k0 * DELTA;
    float t1 = d - (float)k1 * DELTA;
    float g0 = __expf(COEFF * t0 * t0);
    float g1 = (k1 < NGAUSS) ? __expf(COEFF * t1 * t1) : 0.0f;
    ag0[e] = (_Float16)g0;
    ag1[e] = (_Float16)g1;
  }

  // ---- GEMM1: pre = g@Wc + aProj[j] + bProj[i] + bc ; h1 = gelu(pre) -> stage ----
  const size_t rowBi = ((size_t)i * BATCH + b) * HDIM;
#pragma unroll
  for (int nt = 0; nt < 16; ++nt) {
    int h = nt * 16 + l16;                     // output column (C layout: n across lanes)
    const _Float16* pb0 = &sWc[h * 64 + (half << 4)];
    const _Float16* pb1 = &sWc[h * 64 + 32 + (half << 4)];
    v16h bf0 = ld_frag(pb0, pb0 + 8);          // both frags in flight before first wmma
    v16h bf1 = ld_frag(pb1, pb1 + 8);
    // batch global loads up-front so they cluster into one wait
    float brow = bProj[rowBi + h] + bc[h];
    float av[8];
#pragma unroll
    for (int v = 0; v < 8; ++v) {
      int m = v + (half << 3);
      av[v] = aProj[((size_t)(j0 + m) * BATCH + b) * HDIM + h];
    }
    v8f acc = {};
    acc = wmma16(ag0, bf0, acc);
    acc = wmma16(ag1, bf1, acc);
#pragma unroll
    for (int v = 0; v < 8; ++v) {
      int m = v + (half << 3);                 // C layout: row m = vgpr + 8*(lane/16)
      float pre = acc[v] + av[v] + brow;
      stage[m * 256 + h] = (_Float16)gelu_tanh(pre);
    }
  }

  // ---- GEMM2: h2 = h1 + gelu(h1@W_res + b_res) ; restage ----
  v16h af[8];
#pragma unroll
  for (int kb = 0; kb < 8; ++kb) {             // A-frag reload (LDS transpose trick)
    const _Float16* p = &stage[l16 * 256 + kb * 32 + (half << 4)];
    af[kb] = ld_frag(p, p + 8);
  }
#pragma unroll
  for (int nt = 0; nt < 16; ++nt) {
    int h = nt * 16 + l16;
    float br = b_res[h];
    const _Float16* p0 = &sWr[h * 256 + (half << 4)];
    v16h bcur = ld_frag(p0, p0 + 8);
    v8f acc = {};
#pragma unroll
    for (int kb = 0; kb < 8; ++kb) {           // pipeline B-frags one ahead
      v16h bnext = bcur;
      if (kb < 7) {
        const _Float16* p = &sWr[h * 256 + (kb + 1) * 32 + (half << 4)];
        bnext = ld_frag(p, p + 8);
      }
      acc = wmma16(af[kb], bcur, acc);
      bcur = bnext;
    }
#pragma unroll
    for (int v = 0; v < 8; ++v) {
      int m = v + (half << 3);
      float h1 = (float)stage[m * 256 + h];
      float h2 = h1 + gelu_tanh(acc[v] + br);
      stage[m * 256 + h] = (_Float16)h2;       // safe: in-wave LDS ops are in order
    }
  }

  // ---- GEMM3: out = h2@W_out + b_out ; mask ; transposed store ----
  {
    const _Float16* pa0 = &stage[l16 * 256 + (half << 4)];
    const _Float16* pb0 = &sWo[l16 * 256 + (half << 4)];
    v16h acur = ld_frag(pa0, pa0 + 8);
    v16h bcur = ld_frag(pb0, pb0 + 8);
    v8f acc = {};
#pragma unroll
    for (int kb = 0; kb < 8; ++kb) {
      v16h anext = acur, bnext = bcur;
      if (kb < 7) {
        const _Float16* pa = &stage[l16 * 256 + (kb + 1) * 32 + (half << 4)];
        const _Float16* pb = &sWo[l16 * 256 + (kb + 1) * 32 + (half << 4)];
        anext = ld_frag(pa, pa + 8);
        bnext = ld_frag(pb, pb + 8);
      }
      acc = wmma16(acur, bcur, acc);
      acur = anext; bcur = bnext;
    }
    if (l16 < NHEADS) {                        // cols 8..15 are padding
      int n = l16;
      float bo = b_out[n];
      bool mi = mask[b * LNODES + i] != 0;
#pragma unroll
      for (int v = 0; v < 8; ++v) {
        int m = v + (half << 3);
        int j = j0 + m;
        bool mj = mask[b * LNODES + j] != 0;
        float val = acc[v] + bo;
        if (!(mi && mj)) val = -INFINITY;
        // out[(b*NH + n), i, j] of (NM=1, B*NH, L, L)
        out[(((size_t)b * NHEADS + n) * LNODES + i) * LNODES + j] = val;
      }
    }
  }
}

// ---------- launch ----------

extern "C" void kernel_launch(void* const* d_in, const int* in_sizes, int n_in,
                              void* d_out, int out_size, void* d_ws, size_t ws_size,
                              hipStream_t stream) {
  const float* x     = (const float*)d_in[0];
  const float* dist  = (const float*)d_in[1];
  const unsigned char* mask = (const unsigned char*)d_in[2];
  const float* W_rbf = (const float*)d_in[3];
  const float* b_rbf = (const float*)d_in[4];
  const float* W1    = (const float*)d_in[5];
  const float* W2    = (const float*)d_in[6];
  const float* W3    = (const float*)d_in[7];
  const float* b_in  = (const float*)d_in[8];
  const float* W_res = (const float*)d_in[9];
  const float* b_res = (const float*)d_in[10];
  const float* W_out = (const float*)d_in[11];
  const float* b_out = (const float*)d_in[12];
  float* out = (float*)d_out;

  char* ws = (char*)d_ws;
  float*    aP    = (float*)ws;                                   // 1 MB
  float*    bP    = (float*)(ws + (1 << 20));                     // 1 MB
  float*    bc    = (float*)(ws + (2 << 20));                     // 1 KB (pad 4 KB)
  _Float16* WcT   = (_Float16*)(ws + (2 << 20) + 4096);           // 32 KB
  _Float16* WresT = (_Float16*)(ws + (2 << 20) + 4096 + 32768);   // 128 KB
  _Float16* WoutT = (_Float16*)(ws + (2 << 20) + 4096 + 32768 + 131072); // 8 KB

  prep_proj<<<1024, 256, 0, stream>>>(x, W1, W2, aP, bP);
  prep_wc<<<64, 256, 0, stream>>>(W_rbf, W3, b_rbf, b_in, WcT, bc);
  prep_wT<<<256, 256, 0, stream>>>(W_res, W_out, WresT, WoutT);

  // 8192 tiles (128 i * 8 b * 8 j-tiles), 8 waves (tiles) per 256-thread block
  pair_main<<<1024, 256, LDS_BYTES, stream>>>(dist, mask, aP, bP, bc, WcT, WresT,
                                              WoutT, b_res, b_out, out);
}
